// QoRNet_32607391711955
// MI455X (gfx1250) — matrix operand order repaired and checked by reading, not compiled, not run on hardware
//
#include <hip/hip_runtime.h>
#include <hip/hip_bf16.h>
#include <math.h>

#define N_NODES 100000
#define N_PAD   100032   // round up to GEMM tile (64)
#define N_EDGES 600000
#define GG 8
#define HID 128
#define HEADS 4

typedef __attribute__((ext_vector_type(16))) _Float16 v16h;
typedef __attribute__((ext_vector_type(8)))  _Float16 v8h;
typedef __attribute__((ext_vector_type(8)))  float    v8f;
typedef unsigned int u32x4 __attribute__((ext_vector_type(4)));
typedef int          i32x4 __attribute__((ext_vector_type(4)));
typedef int          i32x8 __attribute__((ext_vector_type(8)));

__device__ __forceinline__ float lrelu(float v) { return v > 0.f ? v : 0.2f * v; }

// float atomic max via sign-split integer atomics (valid with -inf init)
__device__ __forceinline__ void atomicMaxF(float* addr, float val) {
    if (val >= 0.f) atomicMax((int*)addr, __float_as_int(val));
    else            atomicMin((unsigned int*)addr, (unsigned int)__float_as_int(val));
}

// ---- Tensor Data Mover: 2D f16 tile (row-major, contiguous) -> LDS with
// pad_enable: +4 DWORDs after every 64 DWORDs => LDS row stride 136 halves
// (272B, keeps 16B alignment for fragment loads, avoids bank conflicts).
__device__ __forceinline__ void tdm_load_2d_f16(
    unsigned lds_addr, const void* gptr,
    unsigned tensor_d0, unsigned tensor_d1, unsigned long long stride0,
    unsigned tile_d0, unsigned tile_d1) {
    const unsigned long long ga = (unsigned long long)gptr;
    u32x4 g0;
    g0[0] = 1u;                                   // count=1, user mode, no gather
    g0[1] = lds_addr;                             // [63:32] LDS byte address
    g0[2] = (unsigned)(ga & 0xFFFFFFFFull);       // global_addr[31:0]
    g0[3] = (unsigned)((ga >> 32) & 0x1FFFFFFull) // global_addr[56:32]
          | (2u << 30);                           // type = 2 ("image")
    i32x8 g1;
    g1[0] = (int)((1u << 16)      // data_size = 1 (2 bytes)
          |       (1u << 20)      // pad_enable
          |       (5u << 22)      // pad_interval: 64 DWORDs
          |       (3u << 25));    // pad_amount: 4 DWORDs
    g1[1] = (int)((tensor_d0 & 0xFFFFu) << 16);
    g1[2] = (int)(((tensor_d0 >> 16) & 0xFFFFu) | ((tensor_d1 & 0xFFFFu) << 16));
    g1[3] = (int)(((tensor_d1 >> 16) & 0xFFFFu) | ((tile_d0 & 0xFFFFu) << 16));
    g1[4] = (int)(tile_d1 & 0xFFFFu);             // tile_dim1; tile_dim2 = 0
    g1[5] = (int)(unsigned)(stride0 & 0xFFFFFFFFull);
    g1[6] = (int)(unsigned)((stride0 >> 32) & 0xFFFFull);  // dim1_stride = 0
    g1[7] = 0;
    const i32x4 z4 = {0, 0, 0, 0};                // 2-D tensor: groups 2/3 unused
    const i32x8 z8 = {0, 0, 0, 0, 0, 0, 0, 0};
    __builtin_amdgcn_tensor_load_to_lds(g0, g1, z4, z4, z8, 0);
}

// ---------------------------------------------------------------- utility
__global__ void k_fill(float* __restrict__ p, float v, long long n) {
    long long i = (long long)blockIdx.x * blockDim.x + threadIdx.x;
    long long s = (long long)gridDim.x * blockDim.x;
    for (; i < n; i += s) p[i] = v;
}

// ------------------------------------------------- node encoder (38 -> 128)
__global__ __launch_bounds__(128) void k_node_encode(
    const float* __restrict__ nn, const int* __restrict__ c0,
    const int* __restrict__ c1, const int* __restrict__ batch,
    const float* __restrict__ recipe, const float* __restrict__ e0,
    const float* __restrict__ e1, const float* __restrict__ W_in,
    const float* __restrict__ b_in, float* __restrict__ h,
    _Float16* __restrict__ hH) {
    __shared__ float f[38];
    const int n = blockIdx.x;
    const int t = threadIdx.x;
    if (t < 8)       f[t] = nn[n * 8 + t];
    else if (t < 14) f[t] = e0[c0[n] * 6 + (t - 8)];
    else if (t < 22) f[t] = e1[c1[n] * 8 + (t - 14)];
    else if (t < 38) f[t] = recipe[batch[n] * 16 + (t - 22)];
    __syncthreads();
    float acc = b_in[t];
#pragma unroll
    for (int k = 0; k < 38; ++k) acc += f[k] * W_in[k * HID + t];
    acc = acc > 0.f ? acc : 0.f;
    h[(long long)n * HID + t]  = acc;
    hH[(long long)n * HID + t] = (_Float16)acc;
}

// -------------------------------------------- in-degree counts (once)
__global__ void k_cnt(const int* __restrict__ dst, float* __restrict__ cnt) {
    int i = blockIdx.x * blockDim.x + threadIdx.x;
    if (i < N_EDGES) atomicAdd(&cnt[dst[i]], 1.f);
}

// ------------- pre-transpose+convert all GAT weights: WtG[l][c][k] (f16)
__global__ __launch_bounds__(128) void k_prep_wt(const float* __restrict__ lin,
                                                 _Float16* __restrict__ WtG) {
    const int b = blockIdx.x;          // 3*HID blocks
    const int l = b >> 7, c = b & 127;
    const int k = threadIdx.x;
    WtG[((size_t)l * HID + c) * HID + k] = (_Float16)lin[((size_t)l * HID + k) * HID + c];
}

// ------------------------ per-layer effective attention matrices (tiny)
__global__ __launch_bounds__(128) void k_weff(
    const float* __restrict__ lin, const float* __restrict__ lin_e,
    const float* __restrict__ att_s, const float* __restrict__ att_d,
    const float* __restrict__ att_e, const float* __restrict__ W_edge,
    const float* __restrict__ b_edge,
    float* __restrict__ weff_s, float* __restrict__ weff_d,
    float* __restrict__ M18, float* __restrict__ bias4) {
    __shared__ float we[HID * HEADS];
    const int k = threadIdx.x;
    float ws[4] = {0, 0, 0, 0}, wd[4] = {0, 0, 0, 0}, wE[4] = {0, 0, 0, 0};
    for (int hc = 0; hc < HID; ++hc) {
        const int hh = hc >> 5;
        const float lv = lin[k * HID + hc];
        const float le = lin_e[k * HID + hc];
        ws[hh] += lv * att_s[hc];
        wd[hh] += lv * att_d[hc];
        wE[hh] += le * att_e[hc];
    }
#pragma unroll
    for (int hh = 0; hh < 4; ++hh) {
        weff_s[k * 4 + hh] = ws[hh];
        weff_d[k * 4 + hh] = wd[hh];
        we[k * 4 + hh]     = wE[hh];
    }
    __syncthreads();
    if (threadIdx.x < 18 * 4) {
        const int i = threadIdx.x >> 2, hh = threadIdx.x & 3;
        float acc = 0.f;
        for (int kk = 0; kk < HID; ++kk) acc += W_edge[i * HID + kk] * we[kk * 4 + hh];
        M18[i * 4 + hh] = acc;
    }
    if (threadIdx.x < 4) {
        float acc = 0.f;
        for (int kk = 0; kk < HID; ++kk) acc += b_edge[kk] * we[kk * 4 + threadIdx.x];
        bias4[threadIdx.x] = acc;
    }
}

// ------------------------------- a_s, a_d = h @ weff (N x 128 x 4 matvec)
__global__ __launch_bounds__(256) void k_node_scores(
    const float* __restrict__ h, const float* __restrict__ weff_s,
    const float* __restrict__ weff_d, float* __restrict__ a_s,
    float* __restrict__ a_d) {
    __shared__ float ws[HID * 4], wd[HID * 4];
    for (int i = threadIdx.x; i < HID * 4; i += 256) { ws[i] = weff_s[i]; wd[i] = weff_d[i]; }
    __syncthreads();
    for (long long n = (long long)blockIdx.x * 256 + threadIdx.x; n < N_NODES;
         n += (long long)gridDim.x * 256) {
        float as[4] = {0, 0, 0, 0}, ad[4] = {0, 0, 0, 0};
        const float* hr = h + n * HID;
        for (int k = 0; k < HID; ++k) {
            const float v = hr[k];
#pragma unroll
            for (int hh = 0; hh < 4; ++hh) { as[hh] += v * ws[k * 4 + hh]; ad[hh] += v * wd[k * 4 + hh]; }
        }
#pragma unroll
        for (int hh = 0; hh < 4; ++hh) { a_s[n * 4 + hh] = as[hh]; a_d[n * 4 + hh] = ad[hh]; }
    }
}

// --------- a_e for real edges: 18-dim gathered feats @ M18 (+ segment sum)
__global__ __launch_bounds__(256) void k_edge_ae(
    const float* __restrict__ en, const int* __restrict__ c0,
    const int* __restrict__ c1, const int* __restrict__ dst,
    const float* __restrict__ emb0, const float* __restrict__ emb1,
    const float* __restrict__ M18, const float* __restrict__ bias4,
    float* __restrict__ ae, float* __restrict__ aesum) {
    __shared__ float M[18 * 4], b4[4];
    for (int i = threadIdx.x; i < 72; i += 256) M[i] = M18[i];
    if (threadIdx.x < 4) b4[threadIdx.x] = bias4[threadIdx.x];
    __syncthreads();
    for (long long j = (long long)blockIdx.x * 256 + threadIdx.x; j < N_EDGES;
         j += (long long)gridDim.x * 256) {
        float f[18];
#pragma unroll
        for (int i = 0; i < 4; ++i) f[i] = en[j * 4 + i];
        const float* g0 = emb0 + (long long)c0[j] * 4;
        const float* g1 = emb1 + (long long)c1[j] * 10;
#pragma unroll
        for (int i = 0; i < 4; ++i) f[4 + i] = g0[i];
#pragma unroll
        for (int i = 0; i < 10; ++i) f[8 + i] = g1[i];
        float acc[4] = {b4[0], b4[1], b4[2], b4[3]};
#pragma unroll
        for (int k = 0; k < 18; ++k) {
            const float v = f[k];
#pragma unroll
            for (int hh = 0; hh < 4; ++hh) acc[hh] += v * M[k * 4 + hh];
        }
        const int d = dst[j];
#pragma unroll
        for (int hh = 0; hh < 4; ++hh) {
            ae[j * 4 + hh] = acc[hh];
            atomicAdd(&aesum[(long long)d * 4 + hh], acc[hh]);
        }
    }
}

// self-loop a_e = segment-mean (linearity of the edge encoder)
__global__ void k_aeloop(const float* __restrict__ aesum, const float* __restrict__ cnt,
                         float* __restrict__ aeloop) {
    long long i = (long long)blockIdx.x * blockDim.x + threadIdx.x;
    if (i < (long long)N_NODES * 4) {
        float c = cnt[i >> 2];
        c = c > 1.f ? c : 1.f;
        aeloop[i] = aesum[i] / c;
    }
}

// ----------- WMMA GEMM: x = hH @ W  (f16 in, TDM-staged LDS, f32 out)
#define GEMM_MT 64
#define ASTR 136   // = 128 + TDM pad (4 DWORDs / row)
#define WSTR 136
__global__ __launch_bounds__(256) void k_gemm_x(
    const _Float16* __restrict__ HH, const _Float16* __restrict__ WtG,
    float* __restrict__ X) {
    __shared__ __align__(16) _Float16 Wt[HID * WSTR];      // B: [col][k]
    __shared__ __align__(16) _Float16 As[GEMM_MT * ASTR];  // A: [row][k]
    const int tid = threadIdx.x;
    const long long row0 = (long long)blockIdx.x * GEMM_MT;
    if (tid < 32) {  // wave 0 drives the Tensor Data Mover
        tdm_load_2d_f16((unsigned)(uintptr_t)&Wt[0], WtG, HID, HID, HID, HID, HID);
        tdm_load_2d_f16((unsigned)(uintptr_t)&As[0], HH + row0 * HID,
                        HID, GEMM_MT, HID, HID, GEMM_MT);
        __builtin_amdgcn_s_wait_tensorcnt(0);
    }
    __syncthreads();
    const int wave = tid >> 5;
    const int lane = tid & 31;
    const int colb = wave * 16;
    const int lhi = lane >> 4;
    const int llo = lane & 15;
    // B fragments: lane holds N=llo, K = lhi*16 + i (contiguous in Wt)
    v16h bfrag[4];
#pragma unroll
    for (int kt = 0; kt < 4; ++kt) {
        const _Float16* bp = &Wt[(colb + llo) * WSTR + kt * 32 + lhi * 16];
        const v8h blo = *(const v8h*)bp;
        const v8h bhi = *(const v8h*)(bp + 8);
#pragma unroll
        for (int i = 0; i < 8; ++i) { bfrag[kt][i] = blo[i]; bfrag[kt][i + 8] = bhi[i]; }
    }
#pragma unroll
    for (int ms = 0; ms < 4; ++ms) {
        v8f acc = {};
#pragma unroll
        for (int kt = 0; kt < 4; ++kt) {
            // A fragment: lane M=llo; K(i) = lhi*16 + i -> two 16B runs
            const _Float16* ap = &As[(ms * 16 + llo) * ASTR + kt * 32 + lhi * 16];
            const v8h alo = *(const v8h*)ap;
            const v8h ahi = *(const v8h*)(ap + 8);
            v16h afrag;
#pragma unroll
            for (int i = 0; i < 8; ++i) { afrag[i] = alo[i]; afrag[i + 8] = ahi[i]; }
            acc = __builtin_amdgcn_wmma_f32_16x16x32_f16(false, afrag, false, bfrag[kt],
                                                         (short)0, acc, false, false);
        }
        // C layout: lanes 0-15 -> M = r, lanes 16-31 -> M = 8+r (per VGPR r)
        float* xp = X + (row0 + ms * 16 + lhi * 8) * HID + (colb + llo);
#pragma unroll
        for (int r = 0; r < 8; ++r) xp[(long long)r * HID] = acc[r];  // rows padded: no guard
    }
}

// ---------------- softmax pass 1: logits + per-dst max (augmented edges)
__global__ void k_logits(const int* __restrict__ src, const int* __restrict__ dst,
                         const float* __restrict__ a_s, const float* __restrict__ a_d,
                         const float* __restrict__ ae, const float* __restrict__ aeloop,
                         float* __restrict__ exbuf, float* __restrict__ dmax) {
    long long j = (long long)blockIdx.x * blockDim.x + threadIdx.x;
    if (j >= (long long)(N_EDGES + N_NODES)) return;
    int s, d;
    const float* aep;
    if (j < N_EDGES) { s = src[j]; d = dst[j]; aep = &ae[j * 4]; }
    else             { s = d = (int)(j - N_EDGES); aep = &aeloop[(long long)d * 4]; }
#pragma unroll
    for (int hh = 0; hh < 4; ++hh) {
        const float lg = lrelu(a_s[(long long)s * 4 + hh] + a_d[(long long)d * 4 + hh] + aep[hh]);
        exbuf[j * 4 + hh] = lg;
        atomicMaxF(&dmax[(long long)d * 4 + hh], lg);
    }
}

// ---------------- softmax pass 2: exp + per-dst sum
__global__ void k_expsum(const int* __restrict__ dst, float* __restrict__ exbuf,
                         const float* __restrict__ dmax, float* __restrict__ dsum) {
    long long j = (long long)blockIdx.x * blockDim.x + threadIdx.x;
    if (j >= (long long)(N_EDGES + N_NODES)) return;
    const int d = (j < N_EDGES) ? dst[j] : (int)(j - N_EDGES);
#pragma unroll
    for (int hh = 0; hh < 4; ++hh) {
        const float v = __expf(exbuf[j * 4 + hh] - dmax[(long long)d * 4 + hh]);
        exbuf[j * 4 + hh] = v;
        atomicAdd(&dsum[(long long)d * 4 + hh], v);
    }
}

// ---------------- weighted scatter-aggregate: out[dst] += alpha * x[src]
__global__ void k_aggr(const int* __restrict__ src, const int* __restrict__ dst,
                       const float* __restrict__ exbuf, const float* __restrict__ dsum,
                       const float* __restrict__ x, float* __restrict__ out) {
    const long long total = (long long)(N_EDGES + N_NODES) * HID;
    long long t = (long long)blockIdx.x * blockDim.x + threadIdx.x;
    if (t >= total) return;
    const long long j = t >> 7;
    const int hc = (int)(t & 127);
    const int hh = hc >> 5;
    int s, d;
    if (j < N_EDGES) { s = src[j]; d = dst[j]; } else { s = d = (int)(j - N_EDGES); }
    const float alpha = exbuf[j * 4 + hh] / dsum[(long long)d * 4 + hh];
    atomicAdd(&out[(long long)d * HID + hc], alpha * x[(long long)s * HID + hc]);
}

__global__ void k_relu_bias(const float* __restrict__ out, const float* __restrict__ bias,
                            float* __restrict__ h, _Float16* __restrict__ hH) {
    long long i = (long long)blockIdx.x * blockDim.x + threadIdx.x;
    if (i < (long long)N_NODES * HID) {
        float v = out[i] + bias[i & 127];
        v = v > 0.f ? v : 0.f;
        h[i]  = v;
        hH[i] = (_Float16)v;
    }
}

__global__ void k_pool(const float* __restrict__ h, const int* __restrict__ batch,
                       float* __restrict__ g) {
    long long i = (long long)blockIdx.x * blockDim.x + threadIdx.x;
    if (i < (long long)N_NODES * HID)
        atomicAdd(&g[(long long)batch[i >> 7] * HID + (i & 127)], h[i]);
}

__global__ __launch_bounds__(128) void k_head(
    const float* __restrict__ g, const float* __restrict__ W1,
    const float* __restrict__ b1, const float* __restrict__ W2,
    const float* __restrict__ b2, float* __restrict__ outp) {
    __shared__ float red[128];
    const int b = blockIdx.x, c = threadIdx.x;
    float acc = b1[c];
    for (int k = 0; k < HID; ++k) acc += g[b * HID + k] * W1[k * HID + c];
    acc = acc > 0.f ? acc : 0.f;
    red[c] = acc * W2[c];
    __syncthreads();
    for (int s = 64; s > 0; s >>= 1) {
        if (c < s) red[c] += red[c + s];
        __syncthreads();
    }
    if (c == 0) outp[b] = red[0] + b2[0];
}

// ---------------------------------------------------------------- launch
extern "C" void kernel_launch(void* const* d_in, const int* in_sizes, int n_in,
                              void* d_out, int out_size, void* d_ws, size_t ws_size,
                              hipStream_t stream) {
    (void)in_sizes; (void)n_in; (void)out_size; (void)ws_size;
    const float* node_numeric = (const float*)d_in[0];
    const int*   node_cat0    = (const int*)d_in[1];
    const int*   node_cat1    = (const int*)d_in[2];
    const float* edge_numeric = (const float*)d_in[3];
    const int*   edge_cat0    = (const int*)d_in[4];
    const int*   edge_cat1    = (const int*)d_in[5];
    const int*   src          = (const int*)d_in[6];
    const int*   dst          = src + N_EDGES;
    const int*   batch        = (const int*)d_in[7];
    const float* recipe       = (const float*)d_in[8];
    const float* emb_node0    = (const float*)d_in[9];
    const float* emb_node1    = (const float*)d_in[10];
    const float* emb_edge0    = (const float*)d_in[11];
    const float* emb_edge1    = (const float*)d_in[12];
    const float* W_in         = (const float*)d_in[13];
    const float* b_in         = (const float*)d_in[14];
    const float* W_edge       = (const float*)d_in[15];
    const float* b_edge       = (const float*)d_in[16];
    const float* gat_lin      = (const float*)d_in[17];
    const float* gat_lin_edge = (const float*)d_in[18];
    const float* att_src      = (const float*)d_in[19];
    const float* att_dst      = (const float*)d_in[20];
    const float* att_edge     = (const float*)d_in[21];
    const float* gat_bias     = (const float*)d_in[22];
    const float* W_r1         = (const float*)d_in[23];
    const float* b_r1         = (const float*)d_in[24];
    const float* W_r2         = (const float*)d_in[25];
    const float* b_r2         = (const float*)d_in[26];

    char* w = (char*)d_ws;
    auto alloc = [&](size_t bytes) {
        void* p = (void*)w;
        w += (bytes + 255) & ~(size_t)255;
        return p;
    };
    float*     h      = (float*)alloc((size_t)N_NODES * HID * 4);
    _Float16*  hH     = (_Float16*)alloc((size_t)N_PAD * HID * 2);  // f16 mirror (padded)
    float*     x      = (float*)alloc((size_t)N_PAD * HID * 4);     // padded: unguarded stores
    float*     outb   = (float*)alloc((size_t)N_NODES * HID * 4);
    _Float16*  WtG    = (_Float16*)alloc((size_t)3 * HID * HID * 2);
    float*     ae     = (float*)alloc((size_t)N_EDGES * 4 * 4);
    float*     exbuf  = (float*)alloc((size_t)(N_EDGES + N_NODES) * 4 * 4);
    float*     a_s    = (float*)alloc((size_t)N_NODES * 4 * 4);
    float*     a_d    = (float*)alloc((size_t)N_NODES * 4 * 4);
    float*     aesum  = (float*)alloc((size_t)N_NODES * 4 * 4);
    float*     aeloop = (float*)alloc((size_t)N_NODES * 4 * 4);
    float*     dmax   = (float*)alloc((size_t)N_NODES * 4 * 4);
    float*     dsum   = (float*)alloc((size_t)N_NODES * 4 * 4);
    float*     cnt    = (float*)alloc((size_t)N_NODES * 4);
    float*     weff_s = (float*)alloc((size_t)HID * 4 * 4);
    float*     weff_d = (float*)alloc((size_t)HID * 4 * 4);
    float*     M18    = (float*)alloc((size_t)18 * 4 * 4);
    float*     bias4  = (float*)alloc((size_t)4 * 4);
    float*     g      = (float*)alloc((size_t)GG * HID * 4);

    const long long NAUG = (long long)N_EDGES + N_NODES;
    const float NEG_INF = -__builtin_huge_valf();

    // zero f16 mirror once (covers the pad rows read by TDM/WMMA)
    k_fill<<<1024, 256, 0, stream>>>((float*)hH, 0.f, (long long)N_PAD * HID / 2);
    k_node_encode<<<N_NODES, 128, 0, stream>>>(node_numeric, node_cat0, node_cat1, batch,
                                               recipe, emb_node0, emb_node1, W_in, b_in,
                                               h, hH);
    k_prep_wt<<<3 * HID, 128, 0, stream>>>(gat_lin, WtG);
    k_fill<<<64, 256, 0, stream>>>(cnt, 0.f, N_NODES);
    k_cnt<<<(N_EDGES + 255) / 256, 256, 0, stream>>>(dst, cnt);

    for (int l = 0; l < 3; ++l) {
        k_weff<<<1, 128, 0, stream>>>(gat_lin + (size_t)l * HID * HID,
                                      gat_lin_edge + (size_t)l * HID * HID,
                                      att_src + l * HID, att_dst + l * HID,
                                      att_edge + l * HID, W_edge, b_edge,
                                      weff_s, weff_d, M18, bias4);
        k_fill<<<2048, 256, 0, stream>>>(outb, 0.f, (long long)N_NODES * HID);
        k_fill<<<64, 256, 0, stream>>>(aesum, 0.f, (long long)N_NODES * 4);
        k_fill<<<64, 256, 0, stream>>>(dmax, NEG_INF, (long long)N_NODES * 4);
        k_fill<<<64, 256, 0, stream>>>(dsum, 0.f, (long long)N_NODES * 4);
        k_node_scores<<<1024, 256, 0, stream>>>(h, weff_s, weff_d, a_s, a_d);
        k_edge_ae<<<1024, 256, 0, stream>>>(edge_numeric, edge_cat0, edge_cat1, dst,
                                            emb_edge0, emb_edge1, M18, bias4, ae, aesum);
        k_aeloop<<<(N_NODES * 4 + 255) / 256, 256, 0, stream>>>(aesum, cnt, aeloop);
        k_gemm_x<<<N_PAD / GEMM_MT, 256, 0, stream>>>(hH, WtG + (size_t)l * HID * HID, x);
        k_logits<<<(int)((NAUG + 255) / 256), 256, 0, stream>>>(src, dst, a_s, a_d, ae,
                                                                aeloop, exbuf, dmax);
        k_expsum<<<(int)((NAUG + 255) / 256), 256, 0, stream>>>(dst, exbuf, dmax, dsum);
        const long long tot = NAUG * HID;
        k_aggr<<<(int)((tot + 255) / 256), 256, 0, stream>>>(src, dst, exbuf, dsum, x, outb);
        k_relu_bias<<<(int)(((long long)N_NODES * HID + 255) / 256), 256, 0, stream>>>(
            outb, gat_bias + l * HID, h, hH);
    }

    k_fill<<<4, 256, 0, stream>>>(g, 0.f, (long long)GG * HID);
    k_pool<<<(int)(((long long)N_NODES * HID + 255) / 256), 256, 0, stream>>>(h, batch, g);
    k_head<<<GG, 128, 0, stream>>>(g, W_r1, b_r1, W_r2, b_r2, (float*)d_out);
}
